// SlotAttention_61108794687651
// MI455X (gfx1250) — compile-verified, matrix operand order checked by reading, not run on hardware
//
#include <hip/hip_runtime.h>

typedef unsigned short u16;
typedef unsigned int   u32;
typedef __attribute__((ext_vector_type(16))) __bf16 v16bf;
typedef __attribute__((ext_vector_type(8)))  float  v8f;

#define BB 64
#define NN 4096
#define SS 8
#define DD 256
#define LN_EPS 1e-5f
#define SA_EPS 1e-8f

__device__ __forceinline__ u16 f2bf(float x) {
  u32 u = __builtin_bit_cast(u32, x);
  u32 r = (u + 0x7FFFu + ((u >> 16) & 1u)) >> 16;
  return (u16)r;
}

union FragU { v16bf v; uint4 q[2]; };

// Load a 16x32 bf16 WMMA operand tile (A, or B from a "rows are the N-dim"
// source) per ISA 7.12.2: lane L -> row r0+(L&15), K chunks of 8 at
// k0+(L>>4)*8 and k0+16+(L>>4)*8.  Two b128 loads per lane.
__device__ __forceinline__ v16bf load_frag(const u16* __restrict__ src, int ld, int r0, int k0) {
  int lane = threadIdx.x & 31;
  int r  = r0 + (lane & 15);
  int kb = k0 + ((lane >> 4) << 3);
  const u16* p = src + (size_t)r * ld + kb;
  FragU f;
  f.q[0] = *(const uint4*)(p);
  f.q[1] = *(const uint4*)(p + 16);
  return f.v;
}

__device__ __forceinline__ v8f wmma_bf16(v16bf a, v16bf b, v8f c) {
  return __builtin_amdgcn_wmma_f32_16x16x32_bf16(false, a, false, b, (short)0, c, false, false);
}

#define V8F_ZERO {0.f,0.f,0.f,0.f,0.f,0.f,0.f,0.f}

// ---------------------------------------------------------------- k0: init
// Convert wk/wv to bf16; zero attn pad rows (s=8..15) for all batches.
__global__ __launch_bounds__(256) void k0_init(const float* __restrict__ wk,
                                               const float* __restrict__ wv,
                                               u16* __restrict__ wk_bf,
                                               u16* __restrict__ wv_bf,
                                               u16* __restrict__ attn_bf) {
  size_t idx = (size_t)blockIdx.x * 256 + threadIdx.x;
  if (idx < (size_t)DD * DD) {
    wk_bf[idx] = f2bf(wk[idx]);
    wv_bf[idx] = f2bf(wv[idx]);
  }
  size_t total = (size_t)BB * 8 * NN;   // grid sized to exactly cover this
  if (idx < total) {
    size_t b = idx / (8 * NN);
    size_t rem = idx % (8 * NN);
    size_t r = rem / NN, n = rem % NN;
    attn_bf[((b * 16) + 8 + r) * NN + n] = 0;
  }
}

// ---------------------------------------------------------------- k1: LN + K/V projection
// Block: 256 thr (8 waves), 64 rows of [B*N].  Phase 1: LN rows -> LDS bf16.
// Phase 2: WMMA x@wk^T and x@wv^T.  k stored row-major bf16 [B][N][D],
// v stored transposed bf16 [B][D][N] via per-wave LDS 16x16 transpose.
__global__ __launch_bounds__(256) void k1_ln_kv(const float* __restrict__ inputs,
                                                const float* __restrict__ g_in,
                                                const float* __restrict__ b_in,
                                                const u16* __restrict__ wk_bf,
                                                const float* __restrict__ bk,
                                                const u16* __restrict__ wv_bf,
                                                const float* __restrict__ bv,
                                                u16* __restrict__ k_out,
                                                u16* __restrict__ vT_out) {
  __shared__ __align__(16) u16 xs[64][DD];        // 32 KB LN'd inputs (bf16)
  __shared__ __align__(16) u16 tbuf[8][16][16];   // 4 KB per-wave transpose scratch
  const int tid = threadIdx.x;
  const int lane = tid & 31, wave = tid >> 5;
  const size_t row0 = (size_t)blockIdx.x * 64;    // global row in [0, B*N)
  const int b = (int)(row0 / NN);
  const int nrow0 = (int)(row0 % NN);

  // ---- Phase 1: LayerNorm, one wave per row (8 rows per wave)
  for (int rr = 0; rr < 8; ++rr) {
    int r = wave * 8 + rr;
    const float* src = inputs + (row0 + r) * DD + lane * 8;
    float x[8];
#pragma unroll
    for (int j = 0; j < 8; ++j) x[j] = src[j];
    float s = 0.f, s2 = 0.f;
#pragma unroll
    for (int j = 0; j < 8; ++j) { s += x[j]; s2 += x[j] * x[j]; }
    for (int off = 1; off < 32; off <<= 1) {
      s  += __shfl_xor(s,  off, 32);
      s2 += __shfl_xor(s2, off, 32);
    }
    float mean = s * (1.0f / DD);
    float var  = s2 * (1.0f / DD) - mean * mean;
    float rinv = rsqrtf(var + LN_EPS);
#pragma unroll
    for (int j = 0; j < 8; ++j) {
      int d = lane * 8 + j;
      float y = (x[j] - mean) * rinv * g_in[d] + b_in[d];
      xs[r][d] = f2bf(y);
    }
  }
  __syncthreads();

  // ---- Phase 2: WMMA.  wave -> row strip (wave&3)*16, col half (wave>>2)
  const int rstrip = (wave & 3) * 16;
  const int rloc = 8 * (lane >> 4);
  const int cl = lane & 15;
  u16* kb_ptr = k_out + ((size_t)b * NN + nrow0) * DD;

  for (int ctl = 0; ctl < 8; ++ctl) {
    int ct = (wave >> 2) * 8 + ctl;
    int c0 = ct * 16;
    v8f ak = V8F_ZERO, av = V8F_ZERO;
#pragma unroll
    for (int kk = 0; kk < 8; ++kk) {
      v16bf a  = load_frag(&xs[0][0], DD, rstrip, kk * 32);   // ds_load_b128
      v16bf bk_f = load_frag(wk_bf, DD, c0, kk * 32);
      v16bf bv_f = load_frag(wv_bf, DD, c0, kk * 32);
      ak = wmma_bf16(a, bk_f, ak);
      av = wmma_bf16(a, bv_f, av);
    }
    int c = c0 + cl;
    float biask = bk[c], biasv = bv[c];
    // k: direct row-major store (16 lanes contiguous per row)
#pragma unroll
    for (int j = 0; j < 8; ++j)
      kb_ptr[(size_t)(rstrip + rloc + j) * DD + c] = f2bf(ak[j] + biask);
    // v: transpose 16x16 tile through per-wave LDS, then b128 stores to vT
#pragma unroll
    for (int j = 0; j < 8; ++j)
      tbuf[wave][cl][rloc + j] = f2bf(av[j] + biasv);
    asm volatile("s_wait_dscnt 0" ::: "memory");
    int dl = lane >> 1, nh = lane & 1;
    uint4 rowv = *(const uint4*)&tbuf[wave][dl][nh * 8];
    *(uint4*)(vT_out + ((size_t)b * DD + (c0 + dl)) * NN +
              (nrow0 + rstrip + nh * 8)) = rowv;
    asm volatile("s_wait_dscnt 0" ::: "memory");
  }
}

// ---------------------------------------------------------------- k2: LN slots + q projection (+ zero sums)
__global__ __launch_bounds__(256) void k2_q(const float* __restrict__ prev,
                                            const float* __restrict__ g_s,
                                            const float* __restrict__ b_s,
                                            const float* __restrict__ wq,
                                            const float* __restrict__ bq,
                                            u16* __restrict__ q_bf,
                                            float* __restrict__ sums) {
  __shared__ float sn[SS][DD];
  __shared__ float mv[SS][2];
  const int b = blockIdx.x, tid = threadIdx.x;
  if (tid < SS) sums[b * SS + tid] = 0.f;
  for (int i = tid; i < SS * DD; i += 256)
    sn[i >> 8][i & 255] = prev[(size_t)b * SS * DD + i];
  __syncthreads();
  if (tid < SS) {
    float s = 0.f, s2 = 0.f;
    for (int d = 0; d < DD; ++d) { float x = sn[tid][d]; s += x; s2 += x * x; }
    float m = s * (1.0f / DD);
    mv[tid][0] = m;
    mv[tid][1] = rsqrtf(s2 * (1.0f / DD) - m * m + LN_EPS);
  }
  __syncthreads();
  for (int i = tid; i < SS * DD; i += 256) {
    int s = i >> 8, d = i & 255;
    sn[s][d] = (sn[s][d] - mv[s][0]) * mv[s][1] * g_s[d] + b_s[d];
  }
  __syncthreads();
  for (int i = tid; i < 16 * DD; i += 256) {
    int s = i >> 8, o = i & 255;
    u16 out = 0;
    if (s < SS) {
      float acc = bq[o];
      const float* wr = wq + (size_t)o * DD;
      for (int d = 0; d < DD; ++d) acc += sn[s][d] * wr[d];
      out = f2bf(acc);
    }
    q_bf[((size_t)b * 16 + s) * DD + o] = out;   // rows 8..15 zero-padded
  }
}

// ---------------------------------------------------------------- k3: dots + softmax(slots) + row sums
// One wave per 16-token tile.  C layout: lane<16 holds dots[s=0..7][n0+lane]
// in its 8 accumulators -> softmax over slots is lane-local.
__global__ __launch_bounds__(256) void k3_dots(const u16* __restrict__ q_bf,
                                               const u16* __restrict__ k_bf,
                                               u16* __restrict__ attn_bf,
                                               float* __restrict__ sums) {
  const int b = blockIdx.x;
  const int wave = threadIdx.x >> 5, lane = threadIdx.x & 31;
  const int nt = blockIdx.y * 8 + wave;
  const int n0 = nt * 16;
  const u16* qb = q_bf + (size_t)b * 16 * DD;
  const u16* kb = k_bf + ((size_t)b * NN + n0) * DD;
  v8f acc = V8F_ZERO;
#pragma unroll
  for (int kk = 0; kk < 8; ++kk) {
    v16bf a = load_frag(qb, DD, 0, kk * 32);
    v16bf bb = load_frag(kb, DD, 0, kk * 32);
    acc = wmma_bf16(a, bb, acc);
  }
  const float scale = 0.0625f;   // 256^-0.5
  float a8[8], m = -3.4e38f;
#pragma unroll
  for (int j = 0; j < 8; ++j) { a8[j] = acc[j] * scale; m = fmaxf(m, a8[j]); }
  float se = 0.f;
#pragma unroll
  for (int j = 0; j < 8; ++j) { a8[j] = __expf(a8[j] - m); se += a8[j]; }
  float inv = 1.0f / se;
#pragma unroll
  for (int j = 0; j < 8; ++j) a8[j] = a8[j] * inv + SA_EPS;
  if (lane < 16) {
    u16* ab = attn_bf + (size_t)b * 16 * NN + n0 + lane;
#pragma unroll
    for (int j = 0; j < 8; ++j) ab[(size_t)j * NN] = f2bf(a8[j]);
  }
  // reduce over the 16 token-lanes (xor network stays inside the 16-group)
  float p[8];
#pragma unroll
  for (int j = 0; j < 8; ++j) p[j] = a8[j];
  for (int off = 1; off < 16; off <<= 1)
#pragma unroll
    for (int j = 0; j < 8; ++j) p[j] += __shfl_xor(p[j], off, 32);
  if (lane == 0)
#pragma unroll
    for (int j = 0; j < 8; ++j) atomicAdd(&sums[b * SS + j], p[j]);
}

// ---------------------------------------------------------------- k4: updates = (attn_un @ v) / rowsum
// Block: (b, d-tile); 8 waves K-split over N=4096; reduce via LDS fp32 tile.
__global__ __launch_bounds__(256) void k4_updates(const u16* __restrict__ attn_bf,
                                                  const u16* __restrict__ vT_bf,
                                                  const float* __restrict__ sums,
                                                  float* __restrict__ upd) {
  __shared__ float tile[16][17];
  const int b = blockIdx.x, dt = blockIdx.y;
  const int tid = threadIdx.x, wave = tid >> 5, lane = tid & 31;
  tile[tid >> 4][tid & 15] = 0.f;
  __syncthreads();
  const u16* ab = attn_bf + (size_t)b * 16 * NN;
  const u16* vb = vT_bf + ((size_t)b * DD + dt * 16) * NN;
  v8f acc = V8F_ZERO;
  const int kbase = wave * 512;
#pragma unroll 4
  for (int kk = 0; kk < 16; ++kk) {
    v16bf a = load_frag(ab, NN, 0, kbase + kk * 32);
    v16bf bb = load_frag(vb, NN, 0, kbase + kk * 32);
    acc = wmma_bf16(a, bb, acc);
  }
  const int c = lane & 15, r0 = 8 * (lane >> 4);
#pragma unroll
  for (int j = 0; j < 8; ++j) atomicAdd(&tile[r0 + j][c], acc[j]);
  __syncthreads();
  if (tid < 128) {
    int s = tid >> 4, c2 = tid & 15;
    float invs = 1.0f / sums[b * SS + s];
    upd[((size_t)b * SS + s) * DD + dt * 16 + c2] = tile[s][c2] * invs;
  }
}

// ---------------------------------------------------------------- k5: GRU cell + residual MLP (fp32 VALU)
__global__ __launch_bounds__(256) void k5_gru_mlp(const float* __restrict__ prev,
                                                  const float* __restrict__ upd,
                                                  const float* __restrict__ w_ih,
                                                  const float* __restrict__ b_ih,
                                                  const float* __restrict__ w_hh,
                                                  const float* __restrict__ b_hh,
                                                  const float* __restrict__ g_ff,
                                                  const float* __restrict__ b_ff,
                                                  const float* __restrict__ w1,
                                                  const float* __restrict__ b1,
                                                  const float* __restrict__ w2,
                                                  const float* __restrict__ b2,
                                                  float* __restrict__ out) {
  __shared__ float P[SS][DD], U[SS][DD], Sn[SS][DD], H[SS][DD];
  __shared__ float mv[SS][2];
  const int b = blockIdx.x, tid = threadIdx.x;
  for (int i = tid; i < SS * DD; i += 256) {
    P[i >> 8][i & 255] = prev[(size_t)b * SS * DD + i];
    U[i >> 8][i & 255] = upd[(size_t)b * SS * DD + i];
  }
  __syncthreads();
  for (int i = tid; i < SS * DD; i += 256) {
    int s = i >> 8, j = i & 255;
    float xr = b_ih[j], xz = b_ih[DD + j], xn = b_ih[2 * DD + j];
    float hr = b_hh[j], hz = b_hh[DD + j], hn = b_hh[2 * DD + j];
    const float* wr = w_ih + (size_t)j * DD;
    const float* wz = w_ih + (size_t)(DD + j) * DD;
    const float* wn = w_ih + (size_t)(2 * DD + j) * DD;
    const float* vr = w_hh + (size_t)j * DD;
    const float* vz = w_hh + (size_t)(DD + j) * DD;
    const float* vn = w_hh + (size_t)(2 * DD + j) * DD;
    for (int d = 0; d < DD; ++d) {
      float u = U[s][d], p = P[s][d];
      xr += u * wr[d]; xz += u * wz[d]; xn += u * wn[d];
      hr += p * vr[d]; hz += p * vz[d]; hn += p * vn[d];
    }
    float r = 1.f / (1.f + __expf(-(xr + hr)));
    float z = 1.f / (1.f + __expf(-(xz + hz)));
    float n = tanhf(xn + r * hn);
    Sn[s][j] = (1.f - z) * n + z * P[s][j];
  }
  __syncthreads();
  if (tid < SS) {
    float s = 0.f, s2 = 0.f;
    for (int d = 0; d < DD; ++d) { float x = Sn[tid][d]; s += x; s2 += x * x; }
    float m = s * (1.0f / DD);
    mv[tid][0] = m;
    mv[tid][1] = rsqrtf(s2 * (1.0f / DD) - m * m + LN_EPS);
  }
  __syncthreads();
  for (int i = tid; i < SS * DD; i += 256) {
    int s = i >> 8, j = i & 255;
    float acc = b1[j];
    const float* wr = w1 + (size_t)j * DD;
    float m = mv[s][0], rv = mv[s][1];
    for (int d = 0; d < DD; ++d)
      acc += ((Sn[s][d] - m) * rv * g_ff[d] + b_ff[d]) * wr[d];
    H[s][j] = fmaxf(acc, 0.f);
  }
  __syncthreads();
  for (int i = tid; i < SS * DD; i += 256) {
    int s = i >> 8, j = i & 255;
    float acc = b2[j];
    const float* wr = w2 + (size_t)j * DD;
    for (int d = 0; d < DD; ++d) acc += H[s][d] * wr[d];
    out[(size_t)b * SS * DD + i] = Sn[s][j] + acc;
  }
}

// ---------------------------------------------------------------- host
extern "C" void kernel_launch(void* const* d_in, const int* in_sizes, int n_in,
                              void* d_out, int out_size, void* d_ws, size_t ws_size,
                              hipStream_t stream) {
  const float* inputs     = (const float*)d_in[0];
  const float* init_slots = (const float*)d_in[1];
  const float* wq  = (const float*)d_in[2];  const float* bq  = (const float*)d_in[3];
  const float* wk  = (const float*)d_in[4];  const float* bk  = (const float*)d_in[5];
  const float* wv  = (const float*)d_in[6];  const float* bv  = (const float*)d_in[7];
  const float* w_ih = (const float*)d_in[8]; const float* b_ih = (const float*)d_in[9];
  const float* w_hh = (const float*)d_in[10];const float* b_hh = (const float*)d_in[11];
  const float* w1  = (const float*)d_in[12]; const float* b1  = (const float*)d_in[13];
  const float* w2  = (const float*)d_in[14]; const float* b2  = (const float*)d_in[15];
  const float* g_in = (const float*)d_in[16];const float* b_in = (const float*)d_in[17];
  const float* g_s  = (const float*)d_in[18];const float* b_s  = (const float*)d_in[19];
  const float* g_ff = (const float*)d_in[20];const float* b_ff = (const float*)d_in[21];

  char* p = (char*)d_ws;
  const size_t SZ_KV = (size_t)BB * NN * DD * 2;          // 134 MB each
  u16* k_bf    = (u16*)p;   p += SZ_KV;
  u16* vT_bf   = (u16*)p;   p += SZ_KV;
  u16* q_bf    = (u16*)p;   p += (size_t)BB * 16 * DD * 2;
  u16* attn_bf = (u16*)p;   p += (size_t)BB * 16 * NN * 2;
  float* sums  = (float*)p; p += (size_t)BB * SS * 4;
  float* upd   = (float*)p; p += (size_t)BB * SS * DD * 4;
  float* slotsA = (float*)p; p += (size_t)BB * SS * DD * 4;
  float* slotsB = (float*)p; p += (size_t)BB * SS * DD * 4;
  u16* wk_bf   = (u16*)p;   p += (size_t)DD * DD * 2;
  u16* wv_bf   = (u16*)p;   p += (size_t)DD * DD * 2;

  k0_init<<<8192, 256, 0, stream>>>(wk, wv, wk_bf, wv_bf, attn_bf);
  k1_ln_kv<<<(BB * NN) / 64, 256, 0, stream>>>(inputs, g_in, b_in, wk_bf, bk,
                                               wv_bf, bv, k_bf, vT_bf);

  const float* prev = init_slots;
  float* outs[3] = {slotsA, slotsB, (float*)d_out};
  for (int it = 0; it < 3; ++it) {
    k2_q<<<BB, 256, 0, stream>>>(prev, g_s, b_s, wq, bq, q_bf, sums);
    k3_dots<<<dim3(BB, NN / 16 / 8), 256, 0, stream>>>(q_bf, k_bf, attn_bf, sums);
    k4_updates<<<dim3(BB, DD / 16), 256, 0, stream>>>(attn_bf, vT_bf, sums, upd);
    k5_gru_mlp<<<BB, 256, 0, stream>>>(prev, upd, w_ih, b_ih, w_hh, b_hh,
                                       g_ff, b_ff, w1, b1, w2, b2, outs[it]);
    prev = outs[it];
  }
}
